// LSTM_12111807774780
// MI455X (gfx1250) — matrix operand order repaired and checked by reading, not compiled, not run on hardware
//
#include <hip/hip_runtime.h>
#include <stdint.h>

// LSTM (B=1024, T=512, H=256, input_size=1) + linear head for MI455X (gfx1250).
// 64 workgroups x 16 batch rows; per timestep: gates[16,1024] = h @ W_hh^T via
// v_wmma_f32_16x16x32_f16 (f16 operands, f32 accumulate). Cell state c lives in
// VGPRs across all 512 steps; h is kept in LDS (f16, A-fragment layout).
// W_hh is pre-converted to f16 (L2-resident, 512 KB) and STREAMED from global
// each step. Anti-hoist: an opaque asm dependence on a uniform zero OFFSET
// (not the pointer) keeps the loads inside the t-loop (no scratch spills)
// while preserving the pointer's global address space (global_load_b128, not
// flat_load_b128 -> W loads use LOADcnt only, decoupled from DScnt).

typedef __attribute__((ext_vector_type(16))) _Float16 v16h;
typedef __attribute__((ext_vector_type(8)))  _Float16 v8h;
typedef __attribute__((ext_vector_type(8)))  float    v8f;

#define B_  1024
#define T_  512
#define H_  256
#define G4  1024   // 4*H

__device__ __forceinline__ float fast_sigmoid(float x) {
    return 1.0f / (1.0f + __expf(-x));
}
__device__ __forceinline__ float fast_tanh(float x) {
    return 1.0f - 2.0f / (__expf(2.0f * x) + 1.0f);
}

// ---------------------------------------------------------------------------
// Prep: W_hh f32 -> f16 (row-major [4H,H]); bias = b_ih + b_hh; W_ih column.
// ---------------------------------------------------------------------------
__global__ __launch_bounds__(256) void lstm_prep(
    const float* __restrict__ whh, const float* __restrict__ wih,
    const float* __restrict__ bih, const float* __restrict__ bhh,
    _Float16* __restrict__ whh16, float* __restrict__ bias,
    float* __restrict__ wihc)
{
    int idx = blockIdx.x * 256 + threadIdx.x;
    if (idx < G4 * H_) whh16[idx] = (_Float16)whh[idx];
    if (idx < G4) {
        bias[idx] = bih[idx] + bhh[idx];
        wihc[idx] = wih[idx];            // W_ih is [4H,1]
    }
}

// ---------------------------------------------------------------------------
// Main: 64 blocks x 256 threads (8 wave32). Block b owns batch rows [16b,16b+16).
// Wave w owns hidden slice [32w, 32w+32) and computes all 4 gates for it:
// 8 C-tiles (16x16 f32) = acc[4][2]. K-loop: 8 x (1 A-frag, 8 B-frags, 8 WMMA).
// ---------------------------------------------------------------------------
__global__ __launch_bounds__(256) void lstm_main(
    const float* __restrict__ x,        // [B, T]
    const _Float16* __restrict__ whh16, // [4H, H] f16 row-major
    const float* __restrict__ bias,     // [4H] = b_ih + b_hh
    const float* __restrict__ wihc,     // [4H]
    const float* __restrict__ wlin,     // [H]
    const float* __restrict__ blin,     // [1]
    float* __restrict__ out)            // [B]
{
    __shared__ _Float16 hS[16][H_];     // h state, f16, row-major (A-frag layout)
    __shared__ float    hF[16][H_];     // final-step h in f32 for the head
    __shared__ float    xcol[16];       // x[:,t] for this batch tile
    __shared__ float    red[16][16];    // head reduction

    const int tid   = threadIdx.x;
    const int wave  = tid >> 5;
    const int lane  = tid & 31;
    const int lhalf = lane >> 4;        // 0: lanes 0-15, 1: lanes 16-31
    const int ncol  = lane & 15;        // N within a 16x16 tile / A-row M
    const int row0  = blockIdx.x * 16;  // first batch row of this WG
    const int j0    = wave * 32;        // hidden slice base for this wave

    // h0 = 0
    for (int i = tid; i < 16 * H_; i += 256) (&hS[0][0])[i] = (_Float16)0.0f;

    // Per-lane gate constants: for C-tile element (m = r+8*lhalf, n = nb+ncol)
    float wih_l[4][2], bias_l[4][2];
    #pragma unroll
    for (int g = 0; g < 4; ++g)
        #pragma unroll
        for (int s = 0; s < 2; ++s) {
            int n = g * H_ + j0 + s * 16 + ncol;
            wih_l[g][s]  = wihc[n];
            bias_l[g][s] = bias[n];
        }

    // Cell state c for this wave's hidden slice, same layout as the C tiles.
    v8f cst[2];
    #pragma unroll
    for (int s = 0; s < 2; ++s)
        #pragma unroll
        for (int r = 0; r < 8; ++r) cst[s][r] = 0.0f;

    if (tid < 16) xcol[tid] = x[(size_t)(row0 + tid) * T_];   // t = 0
    __syncthreads();

    // Per-lane row offset into W (single VGPR); g/s/kk terms become immediate
    // offsets on the global_load_b128s.
    const size_t laneoff = (size_t)(j0 + ncol) * H_ + (size_t)(lhalf * 16);

    // Opaque, wave-uniform zero offset. Redefined by volatile asm each t-step
    // so all W loads are loop-variant (cannot be hoisted + spilled), while the
    // pointer chain stays rooted at the kernel arg (global address space).
    size_t zoff = 0;

    for (int t = 0; t < T_; ++t) {
        asm volatile("" : "+s"(zoff));
        const _Float16* wbase = whh16 + zoff + laneoff;

        v8f acc[4][2];
        #pragma unroll
        for (int g = 0; g < 4; ++g)
            #pragma unroll
            for (int s = 0; s < 2; ++s)
                #pragma unroll
                for (int r = 0; r < 8; ++r) acc[g][s][r] = 0.0f;

        #pragma unroll
        for (int kk = 0; kk < H_; kk += 32) {
            // A fragment (16x32 f16): lane holds row m = ncol,
            // elems 0..15 = K kbase..+15 (kbase = kk + 16*lhalf)
            const int kbase = kk + lhalf * 16;
            v8h alo = *(const v8h*)&hS[ncol][kbase];
            v8h ahi = *(const v8h*)&hS[ncol][kbase + 8];
            v16h A;
            #pragma unroll
            for (int i = 0; i < 8; ++i) { A[i] = alo[i]; A[i + 8] = ahi[i]; }

            if (kk + 32 < H_)   // hint next K-slice of W toward the WGP
                __builtin_prefetch((const void*)(wbase + kk + 32), 0, 1);

            // Batch all 8 B-fragment loads (global, L2-resident W) ...
            v16h Bf[4][2];
            #pragma unroll
            for (int g = 0; g < 4; ++g)
                #pragma unroll
                for (int s = 0; s < 2; ++s)
                    Bf[g][s] = *(const v16h*)(wbase + (size_t)(g * H_ + s * 16) * H_ + kk);

            // ... then the 8 WMMAs for this K-step.
            #pragma unroll
            for (int g = 0; g < 4; ++g)
                #pragma unroll
                for (int s = 0; s < 2; ++s)
                    acc[g][s] = __builtin_amdgcn_wmma_f32_16x16x32_f16(
                        false, A, false, Bf[g][s], (short)0, acc[g][s], false, false);
        }

        // Grab x values for this lane's 8 rows before the write barrier.
        float xv[8];
        #pragma unroll
        for (int r = 0; r < 8; ++r) xv[r] = xcol[r + lhalf * 8];

        __syncthreads();   // all hS reads of step t complete

        // Elementwise gate math in f32; update c in registers; write h (f16).
        #pragma unroll
        for (int s = 0; s < 2; ++s) {
            const int j = j0 + s * 16 + ncol;   // hidden index
            #pragma unroll
            for (int r = 0; r < 8; ++r) {
                const int mm = r + lhalf * 8;   // batch row within tile
                const float xm = xv[r];
                float gi = fast_sigmoid(acc[0][s][r] + xm * wih_l[0][s] + bias_l[0][s]);
                float gf = fast_sigmoid(acc[1][s][r] + xm * wih_l[1][s] + bias_l[1][s]);
                float gg = fast_tanh   (acc[2][s][r] + xm * wih_l[2][s] + bias_l[2][s]);
                float go = fast_sigmoid(acc[3][s][r] + xm * wih_l[3][s] + bias_l[3][s]);
                float cv = gf * cst[s][r] + gi * gg;
                cst[s][r] = cv;
                float hv = go * fast_tanh(cv);
                hS[mm][j] = (_Float16)hv;
                if (t == T_ - 1) hF[mm][j] = hv;
            }
        }

        if (tid < 16 && t + 1 < T_)
            xcol[tid] = x[(size_t)(row0 + tid) * T_ + t + 1];

        __syncthreads();   // hS/xcol for step t+1 visible
    }

    // Linear head: out[row0+m] = dot(hF[m,:], wlin) + blin
    const int orow = tid >> 4;
    const int oc   = tid & 15;
    float p = 0.0f;
    #pragma unroll
    for (int k = 0; k < 16; ++k) p += hF[orow][oc * 16 + k] * wlin[oc * 16 + k];
    red[orow][oc] = p;
    __syncthreads();
    if (tid < 16) {
        float ssum = 0.0f;
        #pragma unroll
        for (int k = 0; k < 16; ++k) ssum += red[tid][k];
        out[row0 + tid] = ssum + blin[0];
    }
}

// ---------------------------------------------------------------------------
extern "C" void kernel_launch(void* const* d_in, const int* in_sizes, int n_in,
                              void* d_out, int out_size, void* d_ws, size_t ws_size,
                              hipStream_t stream) {
    const float* x    = (const float*)d_in[0];  // input_seq [B,T]
    const float* Wih  = (const float*)d_in[1];  // [4H,1]
    const float* Whh  = (const float*)d_in[2];  // [4H,H]
    const float* bih  = (const float*)d_in[3];  // [4H]
    const float* bhh  = (const float*)d_in[4];  // [4H]
    const float* Wlin = (const float*)d_in[5];  // [1,H]
    const float* blin = (const float*)d_in[6];  // [1]
    float* out = (float*)d_out;

    // Scratch layout: [0, 512KB) W_hh f16; then bias f32 (4KB); then W_ih col (4KB).
    _Float16* whh16 = (_Float16*)d_ws;
    float* bias = (float*)((char*)d_ws + (size_t)G4 * H_ * sizeof(_Float16));
    float* wihc = bias + G4;

    lstm_prep<<<(G4 * H_ + 255) / 256, 256, 0, stream>>>(Whh, Wih, bih, bhh,
                                                         whh16, bias, wihc);
    lstm_main<<<B_ / 16, 256, 0, stream>>>(x, whh16, bias, wihc, Wlin, blin, out);
}
